// SoftSmoothAP_30966714204713
// MI455X (gfx1250) — matrix operand order repaired and checked by reading, not compiled, not run on hardware
//
#include <hip/hip_runtime.h>

// SoftSmoothAP for MI455X (gfx1250), wave32.
// B=384 samples, C=48 classes, D=512 dims.
//
//  k0: zero the global accumulator in ws
//  k1: sim = P P^T in fp32 via V_WMMA_F32_16X16X4_F32 (temp=0.01 amplifies sim
//      error 100x -> keep the similarity GEMM fp32)
//  k2: per-i fused kernel: sg = sigmoid(100*(r[q]-r[p])) computed on the fly
//      (never materializing the 226MB fp32 B^3 tensor), contracted against
//      posx = [pos | 1] (384x64, f16) with V_WMMA_F32_16X16X32_F16: one WMMA
//      stream yields both num (cols 0..47) and denom rowsum (col 48).
//      Sigmoid fast path: exp2-domain prescale, v_med3 clamp, v_exp_f32,
//      v_rcp_f32, v_cvt_pk_rtz -> ~5.5 VALU/element.
//  k3: out = 1 - acc/384

typedef __attribute__((ext_vector_type(2)))  float    v2f;
typedef __attribute__((ext_vector_type(8)))  float    v8f;
typedef __attribute__((ext_vector_type(2)))  _Float16 h2;
typedef __attribute__((ext_vector_type(16))) _Float16 v16h;

#define B_N 384
#define C_N 48
#define D_N 512

// 100 / temp scaling folded with log2(e) for exp2-domain sigmoid
#define SIG_SCALE 144.26950408889634f   // 100 * log2(e)
#define SIG_LIM   72.13475204444817f    // 50 * log2(e)

__device__ __forceinline__ float fast_exp2(float x) {
#if __has_builtin(__builtin_amdgcn_exp2f)
  return __builtin_amdgcn_exp2f(x);
#else
  return exp2f(x);   // host-pass-safe fallback; device pass uses the builtin
#endif
}

__device__ __forceinline__ h2 pack_f16(float a, float b) {
  return __builtin_bit_cast(h2, __builtin_amdgcn_cvt_pkrtz(a, b));
}

__global__ __launch_bounds__(1) void init_kernel(float* acc) { acc[0] = 0.0f; }

__global__ __launch_bounds__(1) void finish_kernel(const float* acc, float* out) {
  out[0] = 1.0f - acc[0] * (1.0f / (float)B_N);
}

// ---------------- k1: sim = P P^T (fp32 WMMA 16x16x4) ----------------
// grid (24,24), block 32 (one wave per 16x16 tile of sim).
__global__ __launch_bounds__(32) void sim_kernel(const float* __restrict__ preds,
                                                 float* __restrict__ sim) {
  const int i0 = blockIdx.x * 16, j0 = blockIdx.y * 16;
  const int l = threadIdx.x, lm = l & 15, lh = l >> 4;
  v8f c = {};
#if __has_builtin(__builtin_amdgcn_wmma_f32_16x16x4_f32)
  // fp32 A 16x4 layout: lane l holds (M=lm, K = kb + lh*2 + {0,1}); B mirrors with N=lm.
  for (int kb = 0; kb < D_N; kb += 4) {
    const int k = kb + lh * 2;
    v2f a, b;
    a.x = preds[(i0 + lm) * D_N + k];
    a.y = preds[(i0 + lm) * D_N + k + 1];
    b.x = preds[(j0 + lm) * D_N + k];
    b.y = preds[(j0 + lm) * D_N + k + 1];
    c = __builtin_amdgcn_wmma_f32_16x16x4_f32(false, a, false, b, (short)0, c, false, false);
  }
#else
  for (int r = 0; r < 8; ++r) {
    const int M = i0 + r + 8 * lh, N = j0 + lm;
    float s = 0.f;
    for (int k = 0; k < D_N; ++k) s += preds[M * D_N + k] * preds[N * D_N + k];
    c[r] = s;
  }
#endif
  // C/D layout: lane l, vgpr r -> element (M = r + 8*lh, N = lm)
  for (int r = 0; r < 8; ++r)
    sim[(i0 + r + 8 * lh) * B_N + (j0 + lm)] = c[r];
}

// ---------------- k2: fused SmoothAP per query i ----------------
// grid 384 (one workgroup per i), block 256 (8 waves). Wave w owns p-tiles
// {3w, 3w+1, 3w+2}. K-chunk-outer loop: B-fragments + scaled rq loaded once
// per 32-wide chunk, then 3 A-fragments (16 sigmoids/lane each) feed 12 WMMAs.
__global__ __launch_bounds__(256) void smoothap_kernel(const float* __restrict__ softlabels,
                                                       const float* __restrict__ sim,
                                                       float* __restrict__ acc_out) {
  __shared__ _Float16 posxT[64][B_N];   // [c][q], col 48 = ones, 49..63 = zeros (48KB)
  __shared__ float simrow[B_N];
  __shared__ float pos_div[C_N];
  __shared__ float counts_s[C_N];
  __shared__ float wgsum;

  const int i    = blockIdx.x;
  const int tid  = threadIdx.x;
  const int wid  = tid >> 5;
  const int lane = tid & 31;
  const int lm = lane & 15, lh = lane >> 4;

  // Build posxT (transposed so B-fragments are contiguous 32B LDS reads).
  for (int idx = tid; idx < 64 * B_N; idx += 256) {
    const int c = idx / B_N, q = idx % B_N;
    float v;
    if (c < C_N)       v = (softlabels[q * C_N + c] > 0.0f) ? 1.0f : 0.0f;
    else if (c == C_N) v = 1.0f;   // ones column -> rowsum / denom
    else               v = 0.0f;
    posxT[c][q] = (_Float16)v;
  }
  for (int idx = tid; idx < B_N; idx += 256) simrow[idx] = sim[i * B_N + idx];
  if (tid < C_N) pos_div[tid] = 0.0f;
  if (tid == 0) wgsum = 0.0f;
  __syncthreads();

  if (tid < C_N) {
    float s = 0.0f;
    for (int q = 0; q < B_N; ++q) s += (float)posxT[tid][q];
    counts_s[tid] = s;
  }

  // Pre-scaled row anchors for the 3 owned p-tiles.
  float rps[3];
#pragma unroll
  for (int pt = 0; pt < 3; ++pt)
    rps[pt] = simrow[(wid * 3 + pt) * 16 + lm] * SIG_SCALE;

  v8f acc[3][4];
#pragma unroll
  for (int pt = 0; pt < 3; ++pt)
#pragma unroll
    for (int ct = 0; ct < 4; ++ct) acc[pt][ct] = (v8f){};

  for (int kb = 0; kb < B_N; kb += 32) {
    // B fragments in contiguous k order (same as A) -> lane (lm=N, lh picks K half),
    // element e: K = lh*16 + e. posxT row is 768B, offset 32B-aligned.
    v16h Bt0 = *(const v16h*)&posxT[0 * 16 + lm][kb + lh * 16];
    v16h Bt1 = *(const v16h*)&posxT[1 * 16 + lm][kb + lh * 16];
    v16h Bt2 = *(const v16h*)&posxT[2 * 16 + lm][kb + lh * 16];
    v16h Bt3 = *(const v16h*)&posxT[3 * 16 + lm][kb + lh * 16];

    // Scaled rq values for this lane's 16 K slots (contiguous k order in A:
    // element e -> K = lh*16 + e, matching B). 16B-aligned LDS reads.
    float rqs[16];
    {
      const float4 q0 = *(const float4*)&simrow[kb + lh * 16];
      const float4 q1 = *(const float4*)&simrow[kb + lh * 16 + 4];
      const float4 q2 = *(const float4*)&simrow[kb + lh * 16 + 8];
      const float4 q3 = *(const float4*)&simrow[kb + lh * 16 + 12];
      rqs[0]=q0.x*SIG_SCALE;  rqs[1]=q0.y*SIG_SCALE;  rqs[2]=q0.z*SIG_SCALE;  rqs[3]=q0.w*SIG_SCALE;
      rqs[4]=q1.x*SIG_SCALE;  rqs[5]=q1.y*SIG_SCALE;  rqs[6]=q1.z*SIG_SCALE;  rqs[7]=q1.w*SIG_SCALE;
      rqs[8]=q2.x*SIG_SCALE;  rqs[9]=q2.y*SIG_SCALE;  rqs[10]=q2.z*SIG_SCALE; rqs[11]=q2.w*SIG_SCALE;
      rqs[12]=q3.x*SIG_SCALE; rqs[13]=q3.y*SIG_SCALE; rqs[14]=q3.z*SIG_SCALE; rqs[15]=q3.w*SIG_SCALE;
    }

#pragma unroll
    for (int pt = 0; pt < 3; ++pt) {
      // sg = 1/(1 + 2^clamp((rp-rq)*100*log2e, +-50*log2e)); rcp(2.0)=0.5 exact.
      union { v16h v; h2 p[8]; } A;
#pragma unroll
      for (int e = 0; e < 8; ++e) {
        float x0 = __builtin_amdgcn_fmed3f(rps[pt] - rqs[2 * e],     -SIG_LIM, SIG_LIM);
        float x1 = __builtin_amdgcn_fmed3f(rps[pt] - rqs[2 * e + 1], -SIG_LIM, SIG_LIM);
        const float sg0 = __builtin_amdgcn_rcpf(1.0f + fast_exp2(x0));
        const float sg1 = __builtin_amdgcn_rcpf(1.0f + fast_exp2(x1));
        A.p[e] = pack_f16(sg0, sg1);
      }
      acc[pt][0] = __builtin_amdgcn_wmma_f32_16x16x32_f16(false, A.v, false, Bt0, (short)0, acc[pt][0], false, false);
      acc[pt][1] = __builtin_amdgcn_wmma_f32_16x16x32_f16(false, A.v, false, Bt1, (short)0, acc[pt][1], false, false);
      acc[pt][2] = __builtin_amdgcn_wmma_f32_16x16x32_f16(false, A.v, false, Bt2, (short)0, acc[pt][2], false, false);
      acc[pt][3] = __builtin_amdgcn_wmma_f32_16x16x32_f16(false, A.v, false, Bt3, (short)0, acc[pt][3], false, false);
    }
  }

  // Epilogue per p-tile: denom broadcast + pos_divide partials.
  float csum0 = 0.0f, csum1 = 0.0f, csum2 = 0.0f;
#pragma unroll
  for (int pt = 0; pt < 3; ++pt) {
    const int p0 = (wid * 3 + pt) * 16;
    // denom[p] = 0.5 + rowsum lives in acc[pt][3] column N=0 (lanes lm==0).
    // Lane needs rows r + 8*lh -> broadcast from lane (lane & 16).
    float rden[8];
#pragma unroll
    for (int r = 0; r < 8; ++r) {
      const float d = __shfl(0.5f + acc[pt][3][r], lane & 16, 32);
      rden[r] = 1.0f / d;
    }
#pragma unroll
    for (int ct = 0; ct < 3; ++ct) {
      float s = 0.0f;
#pragma unroll
      for (int r = 0; r < 8; ++r) {
        const int p = p0 + r + 8 * lh;
        const float pv = (float)posxT[ct * 16 + lm][p];
        s += (1.0f + acc[pt][ct][r] - 0.5f * pv) * pv * rden[r];
      }
      if (ct == 0) csum0 += s; else if (ct == 1) csum1 += s; else csum2 += s;
    }
  }

  atomicAdd(&pos_div[0 * 16 + lm], csum0);
  atomicAdd(&pos_div[1 * 16 + lm], csum1);
  atomicAdd(&pos_div[2 * 16 + lm], csum2);
  __syncthreads();

  if (tid < C_N) {
    const float cnt = counts_s[tid];
    float val = 0.0f;
    if (cnt >= 4.0f)   // valid gate; counts>=4 so no div-by-zero
      val = softlabels[i * C_N + tid] * pos_div[tid] / cnt;
    atomicAdd(&wgsum, val);
  }
  __syncthreads();
  if (tid == 0) atomicAdd(acc_out, wgsum);
}

extern "C" void kernel_launch(void* const* d_in, const int* in_sizes, int n_in,
                              void* d_out, int out_size, void* d_ws, size_t ws_size,
                              hipStream_t stream) {
  const float* preds      = (const float*)d_in[0];
  const float* softlabels = (const float*)d_in[1];
  float* out = (float*)d_out;
  float* acc = (float*)d_ws;            // [0]: ap accumulator
  float* sim = (float*)d_ws + 64;       // 384*384 floats (590KB, L2-resident)

  hipLaunchKernelGGL(init_kernel, dim3(1), dim3(1), 0, stream, acc);
  hipLaunchKernelGGL(sim_kernel, dim3(24, 24), dim3(32), 0, stream, preds, sim);
  hipLaunchKernelGGL(smoothap_kernel, dim3(B_N), dim3(256), 0, stream, softlabels, sim, acc);
  hipLaunchKernelGGL(finish_kernel, dim3(1), dim3(1), 0, stream, acc, out);
}
